// SSAttentionKernel_730144440581
// MI455X (gfx1250) — matrix operand order repaired and checked by reading, not compiled, not run on hardware
//
#include <hip/hip_runtime.h>
#include <hip/hip_bf16.h>

// ---------------------------------------------------------------------------
// S4 SSM decode: Cauchy spectrum -> iDFT-as-GEMM -> Toeplitz conv GEMM + dC.
// All GEMMs use v_wmma_f32_16x16x32_f16 (wave32 CDNA5 WMMA).
// ---------------------------------------------------------------------------

#define HH   256     // d_model
#define NN   32      // half state size
#define CC   32      // channels
#define LL   2048    // sequence length
#define LF   1025    // L/2+1
#define KPAD 1040    // padded half-spectrum width (multiple of 16)
#define KDIM 2080    // Re|Im packed K dimension (multiple of 32)
#define BB   2       // batch

typedef __attribute__((ext_vector_type(16))) _Float16 v16h;
typedef __attribute__((ext_vector_type(8)))  _Float16 v8h;
typedef __attribute__((ext_vector_type(8)))  float    v8f;

static __device__ __forceinline__ v8f wmma_f16(v16h a, v16h b, v8f c) {
  return __builtin_amdgcn_wmma_f32_16x16x32_f16(false, a, false, b, (short)0, c,
                                                false, false);
}

// ---------------------------------------------------------------------------
// Kernel 0: iDFT basis (f16).  Bas[k][m], k<KPAD: w_l*cos(2*pi*l*m/L)/L,
// k>=KPAD: -w_l*sin(2*pi*l*m/L)/L ; w_l = 1 at l=0,L/2 else 2; 0 in pad rows.
// Exact integer phase reduction mod 2048 keeps cos/sin args accurate.
// ---------------------------------------------------------------------------
__global__ void basis_fill(_Float16* __restrict__ Bas) {
  int idx = blockIdx.x * blockDim.x + threadIdx.x;
  if (idx >= KDIM * LL) return;
  int k = idx / LL, m = idx % LL;
  int comp = (k >= KPAD) ? 1 : 0;
  int l = comp ? (k - KPAD) : k;
  float v = 0.f;
  if (l <= LL / 2) {
    float wl = (l == 0 || l == LL / 2) ? 1.f : 2.f;
    int ph = (int)(((long long)l * (long long)m) & (LL - 1));
    float th = (float)ph * (6.2831853071795864f / (float)LL);
    v = comp ? (-wl * __sinf(th) * (1.f / (float)LL))
             : ( wl * __cosf(th) * (1.f / (float)LL));
  }
  Bas[idx] = (_Float16)v;
}

// ---------------------------------------------------------------------------
// Kernel 1: zero-padded previous_decode row in f16: PDP[b][h][i], i in [0,4096)
// PDP[i] = (i>=2048) ? pd[i-2048] : 0  (so pd[l-m] = PDP[2048+l-m], causal).
// ---------------------------------------------------------------------------
__global__ void pdp_fill(const float* __restrict__ pd, _Float16* __restrict__ PDP) {
  int idx = blockIdx.x * blockDim.x + threadIdx.x;
  if (idx >= BB * HH * 2 * LL) return;
  int i = idx & (2 * LL - 1);
  int bh = idx >> 12;
  float v = (i >= LL) ? pd[(size_t)bh * LL + (i - LL)] : 0.f;
  PDP[idx] = (_Float16)v;
}

// ---------------------------------------------------------------------------
// Kernel 2: Cauchy spectrum + rank-1 Woodbury.  One thread per (h, l).
// Writes KF[(c*HH+h)*KDIM + l] = Re k_f, [... + KPAD + l] = Im k_f  (f16).
// ---------------------------------------------------------------------------
__global__ void cauchy_spectrum(const float* __restrict__ log_dt,
                                const float* __restrict__ B_ri,
                                const float* __restrict__ P_ri,
                                const float* __restrict__ C_ri,
                                const float* __restrict__ inv_w_real,
                                const float* __restrict__ w_imag,
                                _Float16* __restrict__ KF) {
  int idx = blockIdx.x * blockDim.x + threadIdx.x;
  int h = idx / KPAD;
  int l = idx % KPAD;
  if (h >= HH) return;

  if (l >= LF) {  // zero padding columns 1025..1039 (both Re and Im halves)
    for (int c = 0; c < CC; ++c) {
      _Float16* row = KF + (size_t)(c * HH + h) * KDIM;
      row[l] = (_Float16)0.f;
      row[KPAD + l] = (_Float16)0.f;
    }
    return;
  }

  float dt = __expf(log_dt[h]);
  float th = (float)l * (6.2831853071795864f / (float)LL);
  float ore = __cosf(th), oim = -__sinf(th);          // omega = e^{-i th}
  float dre = 1.f + ore, dim = oim;                   // 1 + omega
  float iden = 1.f / (dre * dre + dim * dim);
  // z = 2(1-omega)*conj(1+omega)/|1+omega|^2
  float ar = 1.f - ore, ai = -oim;
  float zre = 2.f * (ar * dre + ai * dim) * iden;
  float zim = 2.f * (ai * dre - ar * dim) * iden;
  // g = 2/(1+omega)
  float gre = 2.f * dre * iden, gim = -2.f * dim * iden;

  float rr0[CC + 1], ri0[CC + 1], rr1[CC + 1], ri1[CC + 1];
#pragma unroll
  for (int j = 0; j <= CC; ++j) { rr0[j] = 0.f; ri0[j] = 0.f; rr1[j] = 0.f; ri1[j] = 0.f; }

  for (int n = 0; n < NN; ++n) {
    float wr = -__expf(inv_w_real[h * NN + n]) * dt;  // w*dt (real)
    float wi = w_imag[h * NN + n] * dt;               // w*dt (imag)
    // cp = 1/(z - w), cm = 1/(z - conj(w))
    float pr = zre - wr, pi = zim - wi;
    float ps = 1.f / (pr * pr + pi * pi);
    float cpre = pr * ps, cpim = -pi * ps;
    float mr = zre - wr, mi = zim + wi;
    float ms = 1.f / (mr * mr + mi * mi);
    float cmre = mr * ms, cmim = -mi * ms;
    float cr1 = cpre + cmre, cr2 = cmim - cpim;       // t*cp + conj(t)*cm
    float ci1 = cpim + cmim, ci2 = cpre - cmre;

    float Bre = B_ri[(h * NN + n) * 2 + 0], Bim = B_ri[(h * NN + n) * 2 + 1];
    float Pre = P_ri[(h * NN + n) * 2 + 0], Pim = P_ri[(h * NN + n) * 2 + 1];

#pragma unroll
    for (int j = 0; j <= CC; ++j) {
      float cjre, cjim;
      if (j < CC) {
        cjre = C_ri[((size_t)(j * HH + h) * NN + n) * 2 + 0];
        cjim = C_ri[((size_t)(j * HH + h) * NN + n) * 2 + 1];
      } else {            // Q = conj(P)
        cjre = Pre; cjim = -Pim;
      }
      // i = 0 (B row)
      float tre = Bre * cjre - Bim * cjim;
      float tim = Bre * cjim + Bim * cjre;
      rr0[j] += tre * cr1 + tim * cr2;
      ri0[j] += tre * ci1 + tim * ci2;
      // i = 1 (P row)
      tre = Pre * cjre - Pim * cjim;
      tim = Pre * cjim + Pim * cjre;
      rr1[j] += tre * cr1 + tim * cr2;
      ri1[j] += tre * ci1 + tim * ci2;
    }
  }
#pragma unroll
  for (int j = 0; j <= CC; ++j) { rr0[j] *= dt; ri0[j] *= dt; rr1[j] *= dt; ri1[j] *= dt; }

  // Woodbury: k_f[c] = r0[c] - r0[Q]*r1[c]/(1 + r1[Q])
  float qr = 1.f + rr1[CC], qi = ri1[CC];
  float qs = 1.f / (qr * qr + qi * qi);
  float fre = (rr0[CC] * qr + ri0[CC] * qi) * qs;
  float fim = (ri0[CC] * qr - rr0[CC] * qi) * qs;

#pragma unroll
  for (int c = 0; c < CC; ++c) {
    float kre = rr0[c] - (fre * rr1[c] - fim * ri1[c]);
    float kim = ri0[c] - (fre * ri1[c] + fim * rr1[c]);
    float vre = kre * gre - kim * gim;   // * 2/(1+omega)
    float vim = kre * gim + kim * gre;
    _Float16* row = KF + (size_t)(c * HH + h) * KDIM;
    row[l] = (_Float16)vre;
    row[KPAD + l] = (_Float16)vim;
  }
}

// ---------------------------------------------------------------------------
// Kernel 3: iDFT as GEMM.  K16[8192 x 2048] = KF[8192 x 2080] * Bas[2080 x 2048]
// 256 threads / block, tile 128x128.  Basis K-slab staged transposed into a
// double-buffered LDS ping-pong (one s_barrier per K-step).  A sched_barrier
// pins all 16 ds_load_b128 B-operand reads before the 8-WMMA burst so one
// s_wait_dscnt covers the clause and the WMMAs issue back-to-back.
// A k order used here (contiguous, not the ISA table order): lane<16 -> K 0-15 ; lane>=16 -> K 16-31.
// B k order used here (same as A): lane<16 -> K 0-15 ; lane>=16 -> K 16-31; N = lane&15.
// ---------------------------------------------------------------------------
__global__ void __launch_bounds__(256) irfft_gemm(const _Float16* __restrict__ KF,
                                                  const _Float16* __restrict__ Bas,
                                                  _Float16* __restrict__ K16) {
  __shared__ __align__(64) _Float16 lb[2][128 * 32];  // [buf][n][k], 2 x 8 KB
  int m0 = blockIdx.x * 128;
  int n0 = blockIdx.y * 128;
  int wave = threadIdx.x >> 5, lane = threadIdx.x & 31;
  int row = m0 + wave * 16 + (lane & 15);
  const _Float16* arow = KF + (size_t)row * KDIM;
  int ka  = (lane < 16) ? 0 : 8;
  int klo = (lane < 16) ? 0 : 16;

  v8f acc[8] = {};
  int skk = threadIdx.x >> 3;          // 0..31 : K row for staging
  int scs = (threadIdx.x & 7) << 4;    // 0..112: N segment for staging

  int p = 0;
  for (int k0 = 0; k0 < KDIM; k0 += 32, p ^= 1) {
    const _Float16* src = Bas + (size_t)(k0 + skk) * LL + n0 + scs;
    v8h s0 = *(const v8h*)src;
    v8h s1 = *(const v8h*)(src + 8);
    __builtin_prefetch(arow + k0 + 32, 0, 3);
#pragma unroll
    for (int i = 0; i < 8; ++i) {
      lb[p][(scs + i) * 32 + skk]     = s0[i];
      lb[p][(scs + 8 + i) * 32 + skk] = s1[i];
    }
    __syncthreads();   // ping-pong: safe with a single barrier per step

    v16h a;
    *(v8h*)&a       = *(const v8h*)(arow + k0 + klo);
    *((v8h*)&a + 1) = *(const v8h*)(arow + k0 + klo + 8);
    v16h bvv[8];
#pragma unroll
    for (int s = 0; s < 8; ++s)
      bvv[s] = *(const v16h*)&lb[p][(s * 16 + (lane & 15)) * 32 + klo];
    __builtin_amdgcn_sched_barrier(0);   // keep all B loads before WMMA burst
#pragma unroll
    for (int s = 0; s < 8; ++s)
      acc[s] = wmma_f16(a, bvv[s], acc[s]);
    __builtin_amdgcn_sched_barrier(0);
  }

  int nbase = n0 + (lane & 15);
  int mloc = (lane < 16) ? 0 : 8;      // D layout: lanes 16-31 hold rows 8-15
#pragma unroll
  for (int s = 0; s < 8; ++s)
#pragma unroll
    for (int r = 0; r < 8; ++r) {
      int mr = m0 + wave * 16 + mloc + r;
      K16[(size_t)mr * LL + nbase + s * 16] = (_Float16)acc[s][r];
    }
}

// ---------------------------------------------------------------------------
// Kernel 4: causal Toeplitz conv GEMM fused with dC contraction.
// Per (b,h): x[c,l] = sum_{m<=l} k[c,m] * pd[l-m]; out[b,h,l] = sum_c x[c,l]*dC[l,h,c].
// pd row staged in LDS (zero padded), B operand = reversed 16-half LDS window.
// ---------------------------------------------------------------------------
__global__ void __launch_bounds__(256) conv_contract(const _Float16* __restrict__ K16,
                                                     const _Float16* __restrict__ PDP,
                                                     const float* __restrict__ dC,
                                                     float* __restrict__ out) {
  __shared__ __align__(64) _Float16 pl[2 * LL];  // padded pd row, 8 KB
  int l0 = blockIdx.x * 256;
  int h  = blockIdx.y;
  int b  = blockIdx.z;
  int wave = threadIdx.x >> 5, lane = threadIdx.x & 31;

  {
    const _Float16* src = PDP + (size_t)(b * HH + h) * (2 * LL) + threadIdx.x * 16;
    v8h s0 = *(const v8h*)src;
    v8h s1 = *(const v8h*)(src + 8);
    *(v8h*)&pl[threadIdx.x * 16]     = s0;
    *(v8h*)&pl[threadIdx.x * 16 + 8] = s1;
  }
  __syncthreads();

  int j   = lane & 15;
  int ka  = (lane < 16) ? 0 : 8;
  int klo = (lane < 16) ? 0 : 16;
  int lcolw = l0 + wave * 32;
  int c  = lane & 15;
  const _Float16* ar0 = K16 + ((size_t)(c * HH) + h) * LL;
  const _Float16* ar1 = K16 + ((size_t)((c + 16) * HH) + h) * LL;

  v8f acc[2][2] = {};                 // [c-tile][col-subtile]
  int Ktop = l0 + 256;

  for (int m0 = 0; m0 < Ktop; m0 += 32) {
    v16h a0, a1;
    *(v8h*)&a0       = *(const v8h*)(ar0 + m0 + klo);
    *((v8h*)&a0 + 1) = *(const v8h*)(ar0 + m0 + klo + 8);
    *(v8h*)&a1       = *(const v8h*)(ar1 + m0 + klo);
    *((v8h*)&a1 + 1) = *(const v8h*)(ar1 + m0 + klo + 8);
    // Build both reversed B windows first so LDS reads pipeline vs WMMA.
    v16h bv0, bv1;
    {
      int base0 = LL + (lcolw + j) - m0 - klo;       // B[t] = pl[base - t]
      int base1 = base0 + 16;
#pragma unroll
      for (int t = 0; t < 16; ++t) bv0[t] = pl[base0 - t];
#pragma unroll
      for (int t = 0; t < 16; ++t) bv1[t] = pl[base1 - t];
    }
    __builtin_amdgcn_sched_barrier(0);   // loads before the 4-WMMA burst
    acc[0][0] = wmma_f16(a0, bv0, acc[0][0]);
    acc[1][0] = wmma_f16(a1, bv0, acc[1][0]);
    acc[0][1] = wmma_f16(a0, bv1, acc[0][1]);
    acc[1][1] = wmma_f16(a1, bv1, acc[1][1]);
    __builtin_amdgcn_sched_barrier(0);
  }

  // Epilogue: contract over c with dC[l,h,c]; fold lane pairs (p, p^16).
  int cbase = (lane < 16) ? 0 : 8;     // D rows held by this lane
#pragma unroll
  for (int s = 0; s < 2; ++s) {
    int lcol = lcolw + s * 16 + j;
    const float* dcp = dC + ((size_t)lcol * HH + h) * CC;
    float part = 0.f;
#pragma unroll
    for (int r = 0; r < 8; ++r) {
      part += acc[0][s][r] * dcp[cbase + r];
      part += acc[1][s][r] * dcp[16 + cbase + r];
    }
    float tot = part + __shfl_xor(part, 16, 32);
    if (lane < 16) out[((size_t)(b * HH) + h) * LL + lcol] = tot;
  }
}

// ---------------------------------------------------------------------------
// Host launcher.  Workspace layout (bytes, all 256-aligned):
//   KF  @ 0          : 8192*2080*2 = 34,078,720
//   Bas @ 34,078,720 : 2080*2048*2 =  8,519,680
//   K16 @ 42,598,400 : 8192*2048*2 = 33,554,432
//   PDP @ 76,152,832 : 2*256*4096*2 = 4,194,304   (total ~80.3 MB, L2-resident)
// ---------------------------------------------------------------------------
extern "C" void kernel_launch(void* const* d_in, const int* in_sizes, int n_in,
                              void* d_out, int out_size, void* d_ws, size_t ws_size,
                              hipStream_t stream) {
  (void)in_sizes; (void)n_in; (void)out_size; (void)ws_size;
  const float* log_dt = (const float*)d_in[0];
  const float* B_ri   = (const float*)d_in[1];
  const float* P_ri   = (const float*)d_in[2];
  const float* C_ri   = (const float*)d_in[3];
  const float* invw   = (const float*)d_in[4];
  const float* wimag  = (const float*)d_in[5];
  const float* dC     = (const float*)d_in[6];
  const float* pd     = (const float*)d_in[7];
  float* out = (float*)d_out;

  char* ws = (char*)d_ws;
  _Float16* KF  = (_Float16*)(ws);
  _Float16* Bas = (_Float16*)(ws + 34078720u);
  _Float16* K16 = (_Float16*)(ws + 42598400u);
  _Float16* PDP = (_Float16*)(ws + 76152832u);

  basis_fill<<<(KDIM * LL + 255) / 256, 256, 0, stream>>>(Bas);
  pdp_fill<<<(BB * HH * 2 * LL + 255) / 256, 256, 0, stream>>>(pd, PDP);
  cauchy_spectrum<<<(HH * KPAD + 255) / 256, 256, 0, stream>>>(
      log_dt, B_ri, P_ri, C_ri, invw, wimag, KF);
  irfft_gemm<<<dim3(8192 / 128, LL / 128), 256, 0, stream>>>(KF, Bas, K16);
  conv_contract<<<dim3(LL / 256, HH, BB), 256, 0, stream>>>(K16, PDP, dC, out);
}
